// Agent_41987600286198
// MI455X (gfx1250) — matrix operand order repaired and checked by reading, not compiled, not run on hardware
//
#include <hip/hip_runtime.h>
#include <hip/hip_bf16.h>
#include <cmath>

typedef _Float16 half_t;
typedef __attribute__((ext_vector_type(16))) _Float16 v16h;
typedef __attribute__((ext_vector_type(8)))  _Float16 v8h;
typedef __attribute__((ext_vector_type(8)))  float    v8f;

#define TT    128
#define BB    2048
#define OBS_D 128
#define ACT_D 32
#define HID   256
#define DISC  0.97f
#define LAMB  0.95f
#define EPSC  0.2f
#define VFCO  0.5f
#define ENTC  0.01f
#define LOG2PI_F 1.8378770664093453f
#define LOG2_F   0.6931471805599453f

static __device__ __forceinline__ float softplusf(float x) {
    return (x > 20.f) ? x : log1pf(__expf(x));
}

static __device__ __forceinline__ void async_wait0() {
#if __has_builtin(__builtin_amdgcn_s_wait_asynccnt)
    __builtin_amdgcn_s_wait_asynccnt(0);
#else
    asm volatile("s_wait_asynccnt 0x0" ::: "memory");
#endif
}

// ---------------- normalize obs -> f16 ----------------
__global__ void ppo_norm(const float* __restrict__ obs,
                         const float* __restrict__ mean,
                         const float* __restrict__ var,
                         const float* __restrict__ nsteps,
                         half_t* __restrict__ xn, long n)
{
    const float ns1 = nsteps[0] + 1.0f;
    for (long i = (long)blockIdx.x * blockDim.x + threadIdx.x; i < n;
         i += (long)gridDim.x * blockDim.x) {
        int o = (int)(i & (OBS_D - 1));
        float v = var[o] / ns1;
        v = fminf(fmaxf(v, 1e-6f), 1e6f);
        float x = (obs[i] - mean[o]) * rsqrtf(v);
        x = fminf(fmaxf(x, -5.f), 5.f);
        xn[i] = (half_t)x;
    }
}

// ---------------- f32 -> f16 weight convert ----------------
__global__ void ppo_f2h(const float* __restrict__ s, half_t* __restrict__ d, int n)
{
    int i = blockIdx.x * blockDim.x + threadIdx.x;
    if (i < n) d[i] = (half_t)s[i];
}

// ---------------- WMMA GEMM: Y = act(X @ W^T + b) ----------------
// X: rows x KDIM (f16, row-major), W: N x KDIM (f16, row-major)
// Block = 8 waves = 128 rows; each wave: 16 rows x 64 cols; grid.y tiles N by 64.
// Weight tile (64 x KDIM) staged in LDS via async copy (padded rows: +8 halves).
template<int KDIM, bool SILU, bool OUT_HALF>
__global__ __launch_bounds__(256) void ppo_gemm(
    const half_t* __restrict__ X, const half_t* __restrict__ W,
    const float* __restrict__ bias, void* __restrict__ Yv,
    int rows, int N)
{
    constexpr int LROW = KDIM + 8;            // padded LDS row stride (halves)
    __shared__ half_t wlds[64 * LROW];

    const int colBase = blockIdx.y << 6;

    // ---- cooperative async copy of the 64xKDIM weight tile into LDS ----
    {
        const uint64_t gsrc  = (uint64_t)(uintptr_t)(W + (size_t)colBase * KDIM);
        const uint32_t lbase = (uint32_t)(uintptr_t)(&wlds[0]);
        constexpr int CPR    = KDIM / 8;              // 16B chunks per row
        constexpr int NCHUNK = 64 * CPR;              // total 16B chunks
#pragma unroll
        for (int c = 0; c < NCHUNK / 256; ++c) {
            const int idx   = c * 256 + threadIdx.x;
            const int row   = idx / CPR;
            const int inrow = idx - row * CPR;
            const uint64_t ga = gsrc + (uint64_t)idx * 16;
            const uint32_t la = lbase + (uint32_t)(row * LROW * 2 + inrow * 16);
            asm volatile("global_load_async_to_lds_b128 %0, %1, off"
                         :: "v"(la), "v"(ga) : "memory");
        }
        async_wait0();
        __syncthreads();
    }

    const int lane = threadIdx.x & 31;
    const int wave = threadIdx.x >> 5;
    const int rowTile = (blockIdx.x * 8 + wave) << 4;
    const int lmod = lane & 15;
    const int ksel = lane >> 4;                 // 0: K{0..7,16..23}  1: K{8..15,24..31}

    const half_t* __restrict__ xrow = X + (size_t)(rowTile + lmod) * KDIM;
    const half_t* wl = &wlds[lmod * LROW + ksel * 8];

    v8f acc[4] = {};

#pragma unroll
    for (int k0 = 0; k0 < KDIM; k0 += 32) {
        // A fragment: 16 rows x 32 K per wave
        v8h alo = *(const v8h*)(xrow + k0 + ksel * 8);
        v8h ahi = *(const v8h*)(xrow + k0 + 16 + ksel * 8);
        v16h a;
#pragma unroll
        for (int i = 0; i < 8; ++i) { a[i] = alo[i]; a[i + 8] = ahi[i]; }

        // hoist all four B fragments so LDS loads can be claused and the
        // four wmmas issue back-to-back behind a single dscnt wait
        v16h b[4];
#pragma unroll
        for (int j = 0; j < 4; ++j) {
            const half_t* bp = wl + j * 16 * LROW + k0;
            v8h blo = *(const v8h*)(bp);
            v8h bhi = *(const v8h*)(bp + 16);
#pragma unroll
            for (int i = 0; i < 8; ++i) { b[j][i] = blo[i]; b[j][i + 8] = bhi[i]; }
        }
#pragma unroll
        for (int j = 0; j < 4; ++j) {
            acc[j] = __builtin_amdgcn_wmma_f32_16x16x32_f16(
                false, a, false, b[j], (short)0, acc[j], false, false);
        }
    }

#pragma unroll
    for (int j = 0; j < 4; ++j) {
        const int col = colBase + j * 16 + lmod;
        const float bv = bias[col];
#pragma unroll
        for (int r = 0; r < 8; ++r) {
            const int orow = rowTile + r + ksel * 8;
            float v = acc[j][r] + bv;
            if (SILU) v = v / (1.0f + __expf(-v));
            const size_t oi = (size_t)orow * N + col;
            if (OUT_HALF) ((half_t*)Yv)[oi] = (half_t)v;
            else          ((float*)Yv)[oi] = v;
        }
    }
    (void)rows;
}

// ---------------- value head: baseline[i] = h2v[i,:] . vW3 + vB3 ----------------
__global__ __launch_bounds__(256) void ppo_rowdot(
    const half_t* __restrict__ Hm, const float* __restrict__ w,
    const float* __restrict__ b, float* __restrict__ out, int rows)
{
    const int lane = threadIdx.x & 31;
    const int row = blockIdx.x * 8 + (threadIdx.x >> 5);
    if (row >= rows) return;
    const half_t* hr = Hm + (size_t)row * HID;
    float s = 0.f;
#pragma unroll
    for (int k = lane; k < HID; k += 32) s += (float)hr[k] * w[k];
#pragma unroll
    for (int off = 16; off; off >>= 1) s += __shfl_down(s, off);
    if (lane == 0) out[row] = s + b[0];
}

// ---------------- per-(t,b) dist terms: rho, entropy ----------------
__global__ __launch_bounds__(256) void ppo_dist(
    const float* __restrict__ plog, const float* __restrict__ blog,
    const float* __restrict__ action, const float* __restrict__ eps,
    float* __restrict__ rho, float* __restrict__ ent, int n)
{
    int i = blockIdx.x * blockDim.x + threadIdx.x;
    if (i >= n) return;
    const float* pl = plog   + (size_t)i * 2 * ACT_D;
    const float* bl = blog   + (size_t)i * 2 * ACT_D;
    const float* ac = action + (size_t)i * ACT_D;
    const float* ep = eps    + (size_t)i * ACT_D;
    float tlp = 0.f, blp = 0.f, es = 0.f;
#pragma unroll 4
    for (int j = 0; j < ACT_D; ++j) {
        float a   = ac[j];
        float ldj = 2.f * (LOG2_F - a - softplusf(-2.f * a));
        float loc = pl[j];
        float sc  = softplusf(pl[ACT_D + j]) + 0.001f;
        float lsc = logf(sc);
        float z   = (a - loc) / sc;
        tlp += -0.5f * z * z - 0.5f * LOG2PI_F - lsc - ldj;
        float d = loc + sc * ep[j];
        es += 0.5f + 0.5f * LOG2PI_F + lsc + 2.f * (LOG2_F - d - softplusf(-2.f * d));
        float bloc = bl[j];
        float bsc  = softplusf(bl[ACT_D + j]) + 0.001f;
        float zb   = (a - bloc) / bsc;
        blp += -0.5f * zb * zb - 0.5f * LOG2PI_F - logf(bsc) - ldj;
    }
    rho[i] = __expf(tlp - blp);
    ent[i] = es;
}

// ---------------- GAE: one thread per column b, serial over T ----------------
__global__ void ppo_gae(const float* __restrict__ base,
                        const float* __restrict__ reward,
                        const float* __restrict__ done,
                        float* __restrict__ vs, float* __restrict__ adv)
{
    int b = blockIdx.x * blockDim.x + threadIdx.x;
    if (b >= BB) return;
    const float bootstrap = base[(size_t)TT * BB + b];
    float acc = 0.f;
    for (int t = TT - 1; t >= 0; --t) {
        size_t i = (size_t)t * BB + b;
        float d = done[i], r = reward[i], v = base[i];
        float mask = 1.f - d;
        float vtp1 = (t == TT - 1) ? bootstrap : base[i + BB];
        float delta = (r + DISC * (1.f - d) * vtp1 - v) * mask;
        float f = DISC * (1.f - d) * mask * LAMB;
        acc = delta + f * acc;
        vs[i] = acc + v;
    }
    for (int t = 0; t < TT; ++t) {
        size_t i = (size_t)t * BB + b;
        float d = done[i], r = reward[i], v = base[i];
        float mask = 1.f - d;
        float vstp1 = (t == TT - 1) ? bootstrap : vs[i + BB];
        adv[i] = (r + DISC * (1.f - d) * vstp1 - v) * mask;
    }
}

// ---------------- reductions ----------------
__global__ __launch_bounds__(256) void ppo_red1(
    const float* __restrict__ rho, const float* __restrict__ adv,
    const float* __restrict__ vs, const float* __restrict__ base,
    const float* __restrict__ ent,
    float* __restrict__ pp, float* __restrict__ pv, float* __restrict__ pe, int n)
{
    float p = 0.f, v = 0.f, e = 0.f;
    for (int i = blockIdx.x * blockDim.x + threadIdx.x; i < n;
         i += gridDim.x * blockDim.x) {
        float r = rho[i], a = adv[i];
        float s1 = r * a;
        float s2 = fminf(fmaxf(r, 1.f - EPSC), 1.f + EPSC) * a;
        p += fminf(s1, s2);
        float ve = vs[i] - base[i];
        v += ve * ve;
        e += ent[i];
    }
    __shared__ float sp[256], sv[256], se[256];
    int t = threadIdx.x;
    sp[t] = p; sv[t] = v; se[t] = e;
    __syncthreads();
    for (int s = 128; s; s >>= 1) {
        if (t < s) { sp[t] += sp[t + s]; sv[t] += sv[t + s]; se[t] += se[t + s]; }
        __syncthreads();
    }
    if (t == 0) { pp[blockIdx.x] = sp[0]; pv[blockIdx.x] = sv[0]; pe[blockIdx.x] = se[0]; }
}

__global__ __launch_bounds__(512) void ppo_red2(
    const float* __restrict__ pp, const float* __restrict__ pv,
    const float* __restrict__ pe, float* __restrict__ out, int nb, float invN)
{
    __shared__ float sp[512], sv[512], se[512];
    int t = threadIdx.x;
    sp[t] = (t < nb) ? pp[t] : 0.f;
    sv[t] = (t < nb) ? pv[t] : 0.f;
    se[t] = (t < nb) ? pe[t] : 0.f;
    __syncthreads();
    for (int s = 256; s; s >>= 1) {
        if (t < s) { sp[t] += sp[t + s]; sv[t] += sv[t + s]; se[t] += se[t + s]; }
        __syncthreads();
    }
    if (t == 0) {
        float policy_loss  = -(sp[0] * invN);
        float v_loss       = 0.5f * VFCO * (sv[0] * invN);
        float entropy_loss = -ENTC * (se[0] * invN);
        out[0] = policy_loss + v_loss + entropy_loss;
    }
}

extern "C" void kernel_launch(void* const* d_in, const int* in_sizes, int n_in,
                              void* d_out, int out_size, void* d_ws, size_t ws_size,
                              hipStream_t stream)
{
    const float* obs    = (const float*)d_in[0];
    const float* reward = (const float*)d_in[1];
    const float* done   = (const float*)d_in[2];
    const float* logits = (const float*)d_in[3];
    const float* action = (const float*)d_in[4];
    const float* pW1 = (const float*)d_in[5];  const float* pB1 = (const float*)d_in[6];
    const float* pW2 = (const float*)d_in[7];  const float* pB2 = (const float*)d_in[8];
    const float* pW3 = (const float*)d_in[9];  const float* pB3 = (const float*)d_in[10];
    const float* vW1 = (const float*)d_in[11]; const float* vB1 = (const float*)d_in[12];
    const float* vW2 = (const float*)d_in[13]; const float* vB2 = (const float*)d_in[14];
    const float* vW3 = (const float*)d_in[15]; const float* vB3 = (const float*)d_in[16];
    const float* rmean  = (const float*)d_in[17];
    const float* rvar   = (const float*)d_in[18];
    const float* nsteps = (const float*)d_in[19];
    const float* eps    = (const float*)d_in[20];
    (void)in_sizes; (void)n_in; (void)out_size; (void)ws_size;

    constexpr size_t ROWS_P = (size_t)TT * BB;        // 262144
    constexpr size_t ROWS_V = (size_t)(TT + 1) * BB;  // 264192

    char* ws = (char*)d_ws;
    size_t off = 0;
    auto take = [&](size_t bytes) -> char* {
        char* p = ws + off;
        off = (off + bytes + 255) & ~(size_t)255;
        return p;
    };
    half_t* xn   = (half_t*)take(ROWS_V * OBS_D * sizeof(half_t));
    half_t* pW1h = (half_t*)take((size_t)HID * OBS_D * sizeof(half_t));
    half_t* pW2h = (half_t*)take((size_t)HID * HID * sizeof(half_t));
    half_t* pW3h = (half_t*)take((size_t)2 * ACT_D * HID * sizeof(half_t));
    half_t* vW1h = (half_t*)take((size_t)HID * OBS_D * sizeof(half_t));
    half_t* vW2h = (half_t*)take((size_t)HID * HID * sizeof(half_t));
    char*   bufA = take(ROWS_V * HID * sizeof(half_t));
    char*   bufB = take(ROWS_V * HID * sizeof(half_t));
    float*  plog = (float*)bufA;  // alias: h1 dead by the time logits are written
    float*  basev = (float*)take(ROWS_V * sizeof(float));
    float*  rho   = (float*)take(ROWS_P * sizeof(float));
    float*  entb  = (float*)take(ROWS_P * sizeof(float));
    float*  vsb   = (float*)take(ROWS_P * sizeof(float));
    float*  advb  = (float*)take(ROWS_P * sizeof(float));
    float*  pp    = (float*)take(512 * sizeof(float));
    float*  pv    = (float*)take(512 * sizeof(float));
    float*  pe    = (float*)take(512 * sizeof(float));

    // 1) normalize observations to f16
    ppo_norm<<<4096, 256, 0, stream>>>(obs, rmean, rvar, nsteps, xn,
                                       (long)ROWS_V * OBS_D);
    // 2) weights -> f16
    ppo_f2h<<<(HID * OBS_D + 255) / 256, 256, 0, stream>>>(pW1, pW1h, HID * OBS_D);
    ppo_f2h<<<(HID * HID + 255) / 256, 256, 0, stream>>>(pW2, pW2h, HID * HID);
    ppo_f2h<<<(2 * ACT_D * HID + 255) / 256, 256, 0, stream>>>(pW3, pW3h, 2 * ACT_D * HID);
    ppo_f2h<<<(HID * OBS_D + 255) / 256, 256, 0, stream>>>(vW1, vW1h, HID * OBS_D);
    ppo_f2h<<<(HID * HID + 255) / 256, 256, 0, stream>>>(vW2, vW2h, HID * HID);

    // 3) value network (all (T+1)*B rows): WMMA GEMMs + 1-wide head
    ppo_gemm<OBS_D, true, true><<<dim3((unsigned)(ROWS_V / 128), HID / 64), 256, 0, stream>>>(
        xn, vW1h, vB1, bufA, (int)ROWS_V, HID);
    ppo_gemm<HID, true, true><<<dim3((unsigned)(ROWS_V / 128), HID / 64), 256, 0, stream>>>(
        (const half_t*)bufA, vW2h, vB2, bufB, (int)ROWS_V, HID);
    ppo_rowdot<<<(unsigned)(ROWS_V / 8), 256, 0, stream>>>(
        (const half_t*)bufB, vW3, vB3, basev, (int)ROWS_V);

    // 4) policy network (T*B rows): WMMA GEMMs
    ppo_gemm<OBS_D, true, true><<<dim3((unsigned)(ROWS_P / 128), HID / 64), 256, 0, stream>>>(
        xn, pW1h, pB1, bufA, (int)ROWS_P, HID);
    ppo_gemm<HID, true, true><<<dim3((unsigned)(ROWS_P / 128), HID / 64), 256, 0, stream>>>(
        (const half_t*)bufA, pW2h, pB2, bufB, (int)ROWS_P, HID);
    ppo_gemm<HID, false, false><<<dim3((unsigned)(ROWS_P / 128), 1), 256, 0, stream>>>(
        (const half_t*)bufB, pW3h, pB3, plog, (int)ROWS_P, 2 * ACT_D);

    // 5) per-sample distribution terms
    ppo_dist<<<(unsigned)(ROWS_P / 256), 256, 0, stream>>>(
        plog, logits, action, eps, rho, entb, (int)ROWS_P);

    // 6) GAE scan
    ppo_gae<<<BB / 256, 256, 0, stream>>>(basev, reward, done, vsb, advb);

    // 7) final reduction -> scalar loss
    ppo_red1<<<512, 256, 0, stream>>>(rho, advb, vsb, basev, entb, pp, pv, pe,
                                      (int)ROWS_P);
    ppo_red2<<<1, 512, 0, stream>>>(pp, pv, pe, (float*)d_out, 512,
                                    1.0f / (float)ROWS_P);
}
